// SelectiveTokenProcessor_38431367364873
// MI455X (gfx1250) — matrix operand order repaired and checked by reading, not compiled, not run on hardware
//
#include <hip/hip_runtime.h>
#include <math.h>

// ---------------------------------------------------------------------------
// SelectiveTokenProcessor for MI455X (gfx1250, wave32, WMMA + TDM)
// B=4, S=1024, D=1024, H=8 -> N = 4096 tokens, fp32 in/out.
// Weights pre-converted/transposed to bf16 [N][K]; GEMMs use
// v_wmma_f32_16x16x32_bf16 with 32x32 wave tiles; B tiles DMA'd to LDS via
// the Tensor Data Mover when the builtin is available.
// ---------------------------------------------------------------------------

#define NTOK 4096
#define DMODEL 1024
#define NHEAD 8
#define DHEAD 128

typedef __attribute__((ext_vector_type(16))) __bf16 v16bf;
typedef __attribute__((ext_vector_type(8)))  float  v8f;
typedef __attribute__((ext_vector_type(4)))  unsigned int v4u;
typedef __attribute__((ext_vector_type(8)))  int v8i_t;
typedef __attribute__((ext_vector_type(4)))  int v4i_t;

enum { ACT_NONE = 0, ACT_RELU = 1, ACT_GELU = 2 };

#ifndef __has_builtin
#define __has_builtin(x) 0
#endif
#if __has_builtin(__builtin_amdgcn_tensor_load_to_lds) && \
    __has_builtin(__builtin_amdgcn_s_wait_tensorcnt)
#define HAVE_TDM 1
#else
#define HAVE_TDM 0
#endif

__device__ __forceinline__ unsigned short f2bf(float f) {
  // round-to-nearest-even fp32 -> bf16
  unsigned u = __float_as_uint(f);
  unsigned r = u + 0x7FFFu + ((u >> 16) & 1u);
  return (unsigned short)(r >> 16);
}

__device__ __forceinline__ float gelu_exact(float x) {
  return 0.5f * x * (1.0f + erff(x * 0.70710678118654752440f));
}

#if HAVE_TDM
// Issue one TDM load: `rows`=8 rows x 32 bf16 from Wt (row stride K elements)
// into LDS at lds_off, padding 16B after every 64B row -> 80B LDS stride.
__device__ __forceinline__ void tdm_load_rows(unsigned lds_off,
                                              unsigned long long gaddr,
                                              unsigned K, unsigned Ntot) {
  v4u g0;
  g0.x = 1u;                                   // count=1, user descriptor
  g0.y = lds_off;                              // LDS byte address
  g0.z = (unsigned)gaddr;                      // global_addr[31:0]
  g0.w = ((unsigned)(gaddr >> 32) & 0x01FFFFFFu) | (2u << 30);  // [56:32]|type=2
  v8i_t g1;
  // data_size=1(2B), pad_enable=1, pad_interval=3(16 dwords=64B), pad_amount=3(4 dwords=16B)
  g1[0] = (int)((1u << 16) | (1u << 20) | (3u << 22) | (3u << 25));
  g1[1] = (int)((K & 0xFFFFu) << 16);                    // tensor_dim0[15:0] @63:48
  g1[2] = (int)((K >> 16) | ((Ntot & 0xFFFFu) << 16));   // td0[31:16], td1[15:0]
  g1[3] = (int)((Ntot >> 16) | (32u << 16));             // td1[31:16], tile_dim0=32
  g1[4] = 8;                                             // tile_dim1=8 rows, tile_dim2=0
  g1[5] = (int)K;                                        // tensor_dim0_stride[31:0]
  g1[6] = 0;                                             // stride[47:32], td1_stride lo
  g1[7] = 0;
  v4i_t gz = {0, 0, 0, 0};
#if __clang_major__ >= 23
  v8i_t gz8 = {0, 0, 0, 0, 0, 0, 0, 0};
  __builtin_amdgcn_tensor_load_to_lds(g0, g1, gz, gz, gz8, 0);
#else
  __builtin_amdgcn_tensor_load_to_lds(g0, g1, gz, gz, 0);
#endif
}
#endif

// ---------------------------------------------------------------------------
// Weight prep: W[K][N] fp32 -> Wt[N][K] bf16 (tiled 32x32 transpose)
// ---------------------------------------------------------------------------
__global__ void __launch_bounds__(256)
transpose_cvt(const float* __restrict__ in, unsigned short* __restrict__ out,
              int K, int N) {
  __shared__ float t[32][33];
  const int bx = blockIdx.x * 32;   // N
  const int by = blockIdx.y * 32;   // K
  const int c = threadIdx.x & 31, r0 = threadIdx.x >> 5;
#pragma unroll
  for (int r = r0; r < 32; r += 8)
    t[r][c] = in[(size_t)(by + r) * N + bx + c];
  __syncthreads();
#pragma unroll
  for (int r = r0; r < 32; r += 8)
    out[(size_t)(bx + r) * K + by + c] = f2bf(t[c][r]);
}

// ---------------------------------------------------------------------------
// GEMM: C[M,N] = act(A[M,K] @ W[K,N] + bias[N]) (+ res[M,N])
// A fp32 row-major; Wt = bf16 W transposed, [N][K].
// Block tile 128x64, K-step 32, 256 threads = 8 waves (4x2), 32x32 per wave.
// LDS (dynamic): As 128 rows x 80B stride @0 (10240B); Bs 64 rows x 80B @10240.
// ---------------------------------------------------------------------------
#define AS_STRIDE 80
#define BS_OFF 10240
#define GEMM_SMEM 15360

template <int ACT, bool RES>
__global__ void __launch_bounds__(256)
gemm_wmma(const float* __restrict__ A, const unsigned short* __restrict__ Wt,
          const float* __restrict__ bias, const float* __restrict__ res,
          float* __restrict__ C, int M, int N, int K) {
  extern __shared__ char smem[];
  const int tid  = threadIdx.x;
  const int lane = tid & 31;
  const int wave = tid >> 5;
  const int wm2  = wave & 3;      // 32-row slab
  const int wn2  = wave >> 2;     // 32-col slab
  const int tile_m = blockIdx.y * 128;
  const int tile_n = blockIdx.x * 64;

  const int mrow = lane & 15;     // row-in-subtile (A) / col (B,C)
  const int hi   = lane >> 4;
  const int khA2 = hi << 4;       // A K-half byte offset (0/16)
  const int khB2 = hi << 4;       // B K-half byte offset (0/16)

  v8f acc00 = {}, acc01 = {}, acc10 = {}, acc11 = {};

#if HAVE_TDM
  const int rowoff = __builtin_amdgcn_readfirstlane(wave * 8);
#endif

  for (int k0 = 0; k0 < K; k0 += 32) {
    __syncthreads();   // previous tiles fully consumed
#if HAVE_TDM
    {
      unsigned long long ga = (unsigned long long)(const char*)Wt +
          2ull * ((unsigned long long)(unsigned)(tile_n + rowoff) * (unsigned)K +
                  (unsigned)k0);
      tdm_load_rows((unsigned)(BS_OFF + rowoff * AS_STRIDE), ga,
                    (unsigned)K, (unsigned)N);
    }
#else
    {  // manual B staging: 64 rows x 32 bf16, 16B per thread
      int r = tid >> 2, c = (tid & 3) * 8;
      uint4 q = *(const uint4*)(Wt + (size_t)(tile_n + r) * K + k0 + c);
      *(uint4*)(smem + BS_OFF + r * AS_STRIDE + c * 2) = q;
    }
#endif
    // A staging: 128x32 fp32 -> bf16, coalesced float4 loads
#pragma unroll
    for (int ch = 0; ch < 4; ++ch) {
      int e = (ch * 256 + tid) * 4;
      int r = e >> 5, col = e & 31;
      float4 f = *(const float4*)(A + (size_t)(tile_m + r) * K + k0 + col);
      unsigned short* d = (unsigned short*)(smem + r * AS_STRIDE + col * 2);
      d[0] = f2bf(f.x); d[1] = f2bf(f.y); d[2] = f2bf(f.z); d[3] = f2bf(f.w);
    }
#if HAVE_TDM
    __builtin_amdgcn_s_wait_tensorcnt(0);   // own wave's TDM rows landed
#endif
    __syncthreads();                        // all waves' B rows + A stores visible

    // ---- fragments: all aligned b128 LDS loads ----
    union Frag { uint4 q[2]; v16bf v; } fa0, fa1, fb0, fb1;
    const char* aptr0 = smem + (wm2 * 32 + mrow) * AS_STRIDE + khA2;
    const char* aptr1 = aptr0 + 16 * AS_STRIDE;
    fa0.q[0] = *(const uint4*)(aptr0);
    fa0.q[1] = *(const uint4*)(aptr0 + 32);
    fa1.q[0] = *(const uint4*)(aptr1);
    fa1.q[1] = *(const uint4*)(aptr1 + 32);
    const char* bptr0 = smem + BS_OFF + (wn2 * 32 + mrow) * AS_STRIDE + khB2;
    const char* bptr1 = bptr0 + 16 * AS_STRIDE;
    fb0.q[0] = *(const uint4*)(bptr0);
    fb0.q[1] = *(const uint4*)(bptr0 + 32);
    fb1.q[0] = *(const uint4*)(bptr1);
    fb1.q[1] = *(const uint4*)(bptr1 + 32);

    acc00 = __builtin_amdgcn_wmma_f32_16x16x32_bf16(false, fa0.v, false, fb0.v, (short)0, acc00, false, false);
    acc01 = __builtin_amdgcn_wmma_f32_16x16x32_bf16(false, fa0.v, false, fb1.v, (short)0, acc01, false, false);
    acc10 = __builtin_amdgcn_wmma_f32_16x16x32_bf16(false, fa1.v, false, fb0.v, (short)0, acc10, false, false);
    acc11 = __builtin_amdgcn_wmma_f32_16x16x32_bf16(false, fa1.v, false, fb1.v, (short)0, acc11, false, false);
  }

  // ---- epilogue ----
  const int col0 = tile_n + wn2 * 32 + mrow;
  const float b0 = bias[col0];
  const float b1 = bias[col0 + 16];
  const int rb = tile_m + wm2 * 32 + (hi << 3);
#pragma unroll
  for (int e = 0; e < 8; ++e) {
    size_t i0 = (size_t)(rb + e) * N + col0;
    size_t i1 = (size_t)(rb + 16 + e) * N + col0;
    float x00 = acc00[e] + b0, x01 = acc01[e] + b1;
    float x10 = acc10[e] + b0, x11 = acc11[e] + b1;
    if (ACT == ACT_RELU) {
      x00 = fmaxf(x00, 0.f); x01 = fmaxf(x01, 0.f);
      x10 = fmaxf(x10, 0.f); x11 = fmaxf(x11, 0.f);
    }
    if (ACT == ACT_GELU) {
      x00 = gelu_exact(x00); x01 = gelu_exact(x01);
      x10 = gelu_exact(x10); x11 = gelu_exact(x11);
    }
    if (RES) {
      x00 += res[i0]; x01 += res[i0 + 16];
      x10 += res[i1]; x11 += res[i1 + 16];
    }
    C[i0] = x00; C[i0 + 16] = x01;
    C[i1] = x10; C[i1 + 16] = x11;
  }
}

static inline void launch_gemm(const float* A, const unsigned short* Wt,
                               const float* bias, const float* res, float* C,
                               int M, int N, int K, int act, hipStream_t stream) {
  dim3 grid(N / 64, M / 128), block(256);
  if (res) {
    gemm_wmma<ACT_NONE, true><<<grid, block, GEMM_SMEM, stream>>>(A, Wt, bias, res, C, M, N, K);
  } else if (act == ACT_RELU) {
    gemm_wmma<ACT_RELU, false><<<grid, block, GEMM_SMEM, stream>>>(A, Wt, bias, nullptr, C, M, N, K);
  } else if (act == ACT_GELU) {
    gemm_wmma<ACT_GELU, false><<<grid, block, GEMM_SMEM, stream>>>(A, Wt, bias, nullptr, C, M, N, K);
  } else {
    gemm_wmma<ACT_NONE, false><<<grid, block, GEMM_SMEM, stream>>>(A, Wt, bias, nullptr, C, M, N, K);
  }
}

// ---------------------------------------------------------------------------
// Tier assignment
// ---------------------------------------------------------------------------
__global__ void __launch_bounds__(1024)
tier_kernel(const float* __restrict__ aw, const float* __restrict__ th_min,
            const float* __restrict__ th_std, const float* __restrict__ th_enh,
            int* __restrict__ tier) {
  __shared__ float red[1024];
  const int b = blockIdx.x, s = threadIdx.x;
  const float* p = aw + ((size_t)b * 1024 + s) * 20;
  float t = 0.0f;
#pragma unroll
  for (int i = 0; i < 20; ++i) t += p[i];
  red[s] = t;
  __syncthreads();
  for (int k = 512; k > 0; k >>= 1) {
    if (s < k) red[s] = fmaxf(red[s], red[s + k]);
    __syncthreads();
  }
  const float norm = t / (red[0] + 1e-8f);
  const float tmn = 1.0f / (1.0f + expf(-th_min[0]));
  const float tsd = 1.0f / (1.0f + expf(-th_std[0]));
  const float ten = 1.0f / (1.0f + expf(-th_enh[0]));
  tier[b * 1024 + s] = (norm >= ten) ? 3 : (norm >= tsd) ? 2 : (norm >= tmn) ? 1 : 0;
}

__global__ void concat_kernel(const float* __restrict__ pa,
                              const float* __restrict__ pb,
                              float* __restrict__ cc) {
  size_t i = (size_t)blockIdx.x * 256 + threadIdx.x;
  int n = (int)(i >> 11), c = (int)(i & 2047);
  cc[i] = (c < 1024) ? pa[(size_t)n * 1024 + c] : pb[(size_t)n * 1024 + (c - 1024)];
}

// ---------------------------------------------------------------------------
// Premium masked attention (flattened N tokens); K/V per head (2MB) lives in L2.
// ---------------------------------------------------------------------------
__global__ void __launch_bounds__(128)
attn_kernel(const float* __restrict__ qkv, const int* __restrict__ tier,
            float* __restrict__ out) {
  __shared__ float sc[NTOK];
  __shared__ float qsh[DHEAD];
  __shared__ float red[128];
  const int n = blockIdx.x, h = blockIdx.y, t = threadIdx.x;
  const float scale = 0.08838834764831845f;   // 1/sqrt(128)

  qsh[t] = qkv[(size_t)n * 3072 + h * DHEAD + t];
  __syncthreads();

  float lmax = -3.0e38f;
  for (int m = t; m < NTOK; m += 128) {
    float s = -1e9f;
    if (tier[m] == 3) {
      const float* kp = qkv + (size_t)m * 3072 + 1024 + h * DHEAD;
      float d = 0.0f;
#pragma unroll 8
      for (int i = 0; i < DHEAD; ++i) d = fmaf(qsh[i], kp[i], d);
      s = d * scale;
    }
    sc[m] = s;
    lmax = fmaxf(lmax, s);
  }
  red[t] = lmax;
  __syncthreads();
  for (int k = 64; k > 0; k >>= 1) {
    if (t < k) red[t] = fmaxf(red[t], red[t + k]);
    __syncthreads();
  }
  const float mx = red[0];
  __syncthreads();

  float lsum = 0.0f;
  for (int m = t; m < NTOK; m += 128) {
    float e = expf(sc[m] - mx);
    sc[m] = e;
    lsum += e;
  }
  red[t] = lsum;
  __syncthreads();
  for (int k = 64; k > 0; k >>= 1) {
    if (t < k) red[t] += red[t + k];
    __syncthreads();
  }
  const float inv = 1.0f / red[0];

  float acc = 0.0f;
  for (int m = 0; m < NTOK; ++m) {
    float w = sc[m];
    if (w != 0.0f)
      acc = fmaf(w, qkv[(size_t)m * 3072 + 2048 + h * DHEAD + t], acc);
  }
  out[(size_t)n * DMODEL + h * DHEAD + t] = acc * inv;
}

__global__ void combine_kernel(const float* __restrict__ o0, const float* __restrict__ o1,
                               const float* __restrict__ o2, const float* __restrict__ o3,
                               const int* __restrict__ tier, float* __restrict__ out) {
  size_t i = (size_t)blockIdx.x * 256 + threadIdx.x;
  int n = (int)(i >> 10);
  int tr = tier[n];
  const float* src = (tr == 0) ? o0 : (tr == 1) ? o1 : (tr == 2) ? o2 : o3;
  out[i] = src[i];
}

// ---------------------------------------------------------------------------
// Host orchestration
// ---------------------------------------------------------------------------
extern "C" void kernel_launch(void* const* d_in, const int* in_sizes, int n_in,
                              void* d_out, int out_size, void* d_ws, size_t ws_size,
                              hipStream_t stream) {
  (void)in_sizes; (void)n_in; (void)out_size; (void)ws_size;
  const float* te     = (const float*)d_in[0];
  const float* aw     = (const float*)d_in[1];
  const float* th_min = (const float*)d_in[2];
  const float* th_std = (const float*)d_in[3];
  const float* th_enh = (const float*)d_in[4];
  const float* min_w1 = (const float*)d_in[5];
  const float* min_b1 = (const float*)d_in[6];
  const float* min_w2 = (const float*)d_in[7];
  const float* min_b2 = (const float*)d_in[8];
  const float* std_w1 = (const float*)d_in[9];
  const float* std_b1 = (const float*)d_in[10];
  const float* std_w2 = (const float*)d_in[11];
  const float* std_b2 = (const float*)d_in[12];
  const float* ea_w1  = (const float*)d_in[13];
  const float* ea_b1  = (const float*)d_in[14];
  const float* ea_w2  = (const float*)d_in[15];
  const float* ea_b2  = (const float*)d_in[16];
  const float* eb_w1  = (const float*)d_in[17];
  const float* eb_b1  = (const float*)d_in[18];
  const float* eb_w2  = (const float*)d_in[19];
  const float* eb_b2  = (const float*)d_in[20];
  const float* eb_w3  = (const float*)d_in[21];
  const float* eb_b3  = (const float*)d_in[22];
  const float* ec_w1  = (const float*)d_in[23];
  const float* ec_b1  = (const float*)d_in[24];
  const float* ec_w2  = (const float*)d_in[25];
  const float* ec_b2  = (const float*)d_in[26];
  const float* pf_w1  = (const float*)d_in[27];
  const float* pf_b1  = (const float*)d_in[28];
  const float* pf_w2  = (const float*)d_in[29];
  const float* pf_b2  = (const float*)d_in[30];
  const float* wqkv   = (const float*)d_in[31];
  const float* bqkv   = (const float*)d_in[32];
  const float* wo     = (const float*)d_in[33];
  const float* bo     = (const float*)d_in[34];
  const float* pp_w1  = (const float*)d_in[35];
  const float* pp_b1  = (const float*)d_in[36];
  const float* pp_w2  = (const float*)d_in[37];
  const float* pp_b2  = (const float*)d_in[38];

  char* ws = (char*)d_ws;
  size_t off = 0;
  auto alloc = [&](size_t bytes) -> void* {
    void* p = ws + off;
    off = (off + bytes + 255) & ~(size_t)255;
    return p;
  };
  int*   tier   = (int*)  alloc((size_t)NTOK * sizeof(int));
  float* o0     = (float*)alloc((size_t)NTOK * 1024 * 4);
  float* o1     = (float*)alloc((size_t)NTOK * 1024 * 4);
  float* o2     = (float*)alloc((size_t)NTOK * 1024 * 4);
  float* o3     = (float*)alloc((size_t)NTOK * 1024 * 4);
  float* h512a  = (float*)alloc((size_t)NTOK * 512 * 4);
  float* h512b  = (float*)alloc((size_t)NTOK * 512 * 4);
  float* hA     = (float*)alloc((size_t)NTOK * 1024 * 4);
  float* hB     = (float*)alloc((size_t)NTOK * 1024 * 4);
  float* hP     = (float*)alloc((size_t)NTOK * 1024 * 4);
  float* h2048  = (float*)alloc((size_t)NTOK * 2048 * 4);
  float* qkvbuf = (float*)alloc((size_t)NTOK * 3072 * 4);
  float* aout   = (float*)alloc((size_t)NTOK * 1024 * 4);
  // transposed bf16 weights
  typedef unsigned short u16;
  u16* t_min_w1 = (u16*)alloc((size_t)1024 * 512 * 2);
  u16* t_min_w2 = (u16*)alloc((size_t)512 * 1024 * 2);
  u16* t_std_w1 = (u16*)alloc((size_t)1024 * 1024 * 2);
  u16* t_std_w2 = (u16*)alloc((size_t)1024 * 1024 * 2);
  u16* t_ea_w1  = (u16*)alloc((size_t)1024 * 1024 * 2);
  u16* t_ea_w2  = (u16*)alloc((size_t)1024 * 1024 * 2);
  u16* t_eb_w1  = (u16*)alloc((size_t)1024 * 512 * 2);
  u16* t_eb_w2  = (u16*)alloc((size_t)512 * 512 * 2);
  u16* t_eb_w3  = (u16*)alloc((size_t)512 * 1024 * 2);
  u16* t_ec_w1  = (u16*)alloc((size_t)2048 * 1024 * 2);
  u16* t_ec_w2  = (u16*)alloc((size_t)1024 * 1024 * 2);
  u16* t_pf_w1  = (u16*)alloc((size_t)3 * 1024 * 2048 * 2);
  u16* t_pf_w2  = (u16*)alloc((size_t)3 * 2048 * 1024 * 2);
  u16* t_wqkv   = (u16*)alloc((size_t)1024 * 3072 * 2);
  u16* t_wo     = (u16*)alloc((size_t)1024 * 1024 * 2);
  u16* t_pp_w1  = (u16*)alloc((size_t)1024 * 1024 * 2);
  u16* t_pp_w2  = (u16*)alloc((size_t)1024 * 1024 * 2);

  auto tr = [&](const float* w, u16* wt, int K, int N) {
    transpose_cvt<<<dim3(N / 32, K / 32), 256, 0, stream>>>(w, wt, K, N);
  };
  tr(min_w1, t_min_w1, 1024, 512);
  tr(min_w2, t_min_w2, 512, 1024);
  tr(std_w1, t_std_w1, 1024, 1024);
  tr(std_w2, t_std_w2, 1024, 1024);
  tr(ea_w1,  t_ea_w1,  1024, 1024);
  tr(ea_w2,  t_ea_w2,  1024, 1024);
  tr(eb_w1,  t_eb_w1,  1024, 512);
  tr(eb_w2,  t_eb_w2,  512, 512);
  tr(eb_w3,  t_eb_w3,  512, 1024);
  tr(ec_w1,  t_ec_w1,  2048, 1024);
  tr(ec_w2,  t_ec_w2,  1024, 1024);
  for (int i = 0; i < 3; ++i) {
    tr(pf_w1 + (size_t)i * 1024 * 2048, t_pf_w1 + (size_t)i * 2048 * 1024, 1024, 2048);
    tr(pf_w2 + (size_t)i * 2048 * 1024, t_pf_w2 + (size_t)i * 1024 * 2048, 2048, 1024);
  }
  tr(wqkv, t_wqkv, 1024, 3072);
  tr(wo,   t_wo,   1024, 1024);
  tr(pp_w1, t_pp_w1, 1024, 1024);
  tr(pp_w2, t_pp_w2, 1024, 1024);

  tier_kernel<<<4, 1024, 0, stream>>>(aw, th_min, th_std, th_enh, tier);

  const int M = NTOK;
  // tier 0
  launch_gemm(te, t_min_w1, min_b1, nullptr, h512a, M, 512, 1024, ACT_RELU, stream);
  launch_gemm(h512a, t_min_w2, min_b2, nullptr, o0, M, 1024, 512, ACT_NONE, stream);
  // tier 1
  launch_gemm(te, t_std_w1, std_b1, nullptr, hA, M, 1024, 1024, ACT_GELU, stream);
  launch_gemm(hA, t_std_w2, std_b2, nullptr, o1, M, 1024, 1024, ACT_NONE, stream);
  // tier 2 (pa in hB, pb in hA)
  launch_gemm(te, t_ea_w1, ea_b1, nullptr, hA, M, 1024, 1024, ACT_GELU, stream);
  launch_gemm(hA, t_ea_w2, ea_b2, nullptr, hB, M, 1024, 1024, ACT_NONE, stream);
  launch_gemm(te, t_eb_w1, eb_b1, nullptr, h512a, M, 512, 1024, ACT_GELU, stream);
  launch_gemm(h512a, t_eb_w2, eb_b2, nullptr, h512b, M, 512, 512, ACT_GELU, stream);
  launch_gemm(h512b, t_eb_w3, eb_b3, nullptr, hA, M, 1024, 512, ACT_NONE, stream);
  concat_kernel<<<(NTOK * 2048) / 256, 256, 0, stream>>>(hB, hA, h2048);
  launch_gemm(h2048, t_ec_w1, ec_b1, nullptr, hA, M, 1024, 2048, ACT_GELU, stream);
  launch_gemm(hA, t_ec_w2, ec_b2, nullptr, o2, M, 1024, 1024, ACT_NONE, stream);
  // tier 3: residual FFN stack (te -> hA -> hB -> hP)
  launch_gemm(te, t_pf_w1 + 0 * 2048 * 1024, pf_b1 + 0 * 2048, nullptr, h2048, M, 2048, 1024, ACT_GELU, stream);
  launch_gemm(h2048, t_pf_w2 + 0 * 1024 * 2048, pf_b2 + 0 * 1024, te, hA, M, 1024, 2048, ACT_NONE, stream);
  launch_gemm(hA, t_pf_w1 + 1 * 2048 * 1024, pf_b1 + 1 * 2048, nullptr, h2048, M, 2048, 1024, ACT_GELU, stream);
  launch_gemm(h2048, t_pf_w2 + 1 * 1024 * 2048, pf_b2 + 1 * 1024, hA, hB, M, 1024, 2048, ACT_NONE, stream);
  launch_gemm(hB, t_pf_w1 + 2 * 2048 * 1024, pf_b1 + 2 * 2048, nullptr, h2048, M, 2048, 1024, ACT_GELU, stream);
  launch_gemm(h2048, t_pf_w2 + 2 * 1024 * 2048, pf_b2 + 2 * 1024, hB, hP, M, 1024, 2048, ACT_NONE, stream);
  // attention
  launch_gemm(hP, t_wqkv, bqkv, nullptr, qkvbuf, M, 3072, 1024, ACT_NONE, stream);
  attn_kernel<<<dim3(NTOK, NHEAD), 128, 0, stream>>>(qkvbuf, tier, aout);
  launch_gemm(aout, t_wo, bo, nullptr, hA, M, 1024, 1024, ACT_NONE, stream);
  launch_gemm(hA, t_pp_w1, pp_b1, nullptr, hB, M, 1024, 1024, ACT_GELU, stream);
  launch_gemm(hB, t_pp_w2, pp_b2, nullptr, o3, M, 1024, 1024, ACT_NONE, stream);
  // combine
  combine_kernel<<<(NTOK * DMODEL) / 256, 256, 0, stream>>>(o0, o1, o2, o3, tier, (float*)d_out);
}
